// FullNN_50998441672942
// MI455X (gfx1250) — matrix operand order repaired and checked by reading, not compiled, not run on hardware
//
#include <hip/hip_runtime.h>
#include <hip/hip_bf16.h>

typedef __attribute__((ext_vector_type(16))) _Float16 v16h;
typedef __attribute__((ext_vector_type(8)))  float    v8f;

#define E_ 3
#define N_ 100000
#define P_ 36
#define H_ 64
#define B_ 2000
#define Q_ (E_ * N_)
#define NNZ_ 20000000
#define CH_ 1024
#define NCHUNK_ ((Q_ + CH_ - 1) / CH_)     // 293
#define TILES_PER_E_ ((N_ + 127) / 128)    // 782

// ---- WMMA fragment index helpers (CDNA5 ISA 7.12.2, wave32) ----
// 16-bit A matrix 16x32: lane L holds row M=L%16; half h of the v16h maps to
// K = (vgpr/4)*16 + (L>=16 ? 8 : 0) + (vgpr%4)*2 + (h&1), vgpr = h/2.
__device__ __forceinline__ int a_klocal(int h, int hi) {
  int i = h >> 1, lo = h & 1;
  return ((i >> 2) << 4) + (hi ? 8 : 0) + ((i & 3) << 1) + lo;
}
// 16-bit B matrix 32x16: lane L holds column N=L%16; half j maps to
// K = (j/8)*16 + (L>=16 ? 8 : 0) + j%8.
__device__ __forceinline__ int b_klocal(int j, int hi) {
  return ((j >> 3) << 4) + (hi ? 8 : 0) + (j & 7);
}

// =====================================================================
// Fused forward + backward MLP. One wave = 16 atoms, 8 waves/block.
// Writes per-atom energy contribution (atomicAdd into energy[idx]) and
// dE/dx [Q, P] into workspace.
// =====================================================================
__global__ __launch_bounds__(256) void mlp_fwd_bwd(
    const float* __restrict__ x,  const float* __restrict__ W0,
    const float* __restrict__ b0, const float* __restrict__ W1,
    const float* __restrict__ b1, const float* __restrict__ W2,
    const float* __restrict__ b2, const int*   __restrict__ idx,
    float* __restrict__ energy,   float* __restrict__ dE) {
  __shared__ float    sW0[H_ * P_];     // [h][p], stride 36 (conflict-free)
  __shared__ float    sW1[H_ * 66];     // [h][k], stride 66 (conflict-free)
  __shared__ float    sW2[H_];
  __shared__ _Float16 sH0[8 * 16 * 72]; // per-wave h0, f16, stride 72 halves
  __shared__ _Float16 sG [8 * 16 * 72]; // per-wave h1 -> g1 -> g0

  const int e    = blockIdx.y;
  const int base = blockIdx.x * 128;
  const int tid  = threadIdx.x;
  const int wave = tid >> 5;
  const int lane = tid & 31;
  const int m    = lane & 15;      // A-row / B-column / D-column index
  const int hi   = lane >> 4;      // lane-half

  for (int i = tid; i < H_ * P_; i += 256) sW0[i] = W0[e * H_ * P_ + i];
  for (int i = tid; i < H_ * H_; i += 256)
    sW1[(i >> 6) * 66 + (i & 63)] = W1[e * H_ * H_ + i];
  if (tid < H_) sW2[tid] = W2[e * H_ + tid];
  __syncthreads();

  _Float16* myH0 = &sH0[wave * 16 * 72];
  _Float16* myG  = &sG [wave * 16 * 72];
  const int tileAtom = base + wave * 16;   // first atom of this wave's tile
  const int aRow = tileAtom + m;           // atom supplying this lane's A rows

  v8f acc[4];

  // ---------------- layer 0: h0 = tanh(x @ W0^T + b0) ----------------
#pragma unroll
  for (int t = 0; t < 4; t++) {
    float bv = b0[e * H_ + m + 16 * t];
#pragma unroll
    for (int v = 0; v < 8; v++) acc[t][v] = bv;
  }
#pragma unroll
  for (int c = 0; c < 2; c++) {
    v16h a = {};
#pragma unroll
    for (int h = 0; h < 16; h++) {
      int k = c * 32 + a_klocal(h, hi);
      float val = 0.f;
      if (k < P_ && aRow < N_) val = x[(e * N_ + aRow) * P_ + k];
      a[h] = (_Float16)val;
    }
#pragma unroll
    for (int t = 0; t < 4; t++) {
      v16h bf = {};
#pragma unroll
      for (int j = 0; j < 16; j++) {
        int k = c * 32 + b_klocal(j, hi);                 // p index
        bf[j] = (_Float16)((k < P_) ? sW0[(m + 16 * t) * P_ + k] : 0.f);
      }
      acc[t] = __builtin_amdgcn_wmma_f32_16x16x32_f16(
          false, a, false, bf, (short)0, acc[t], false, false);
    }
  }
#pragma unroll
  for (int t = 0; t < 4; t++)
#pragma unroll
    for (int v = 0; v < 8; v++) {
      int row = v + 8 * hi, col = m + 16 * t;
      myH0[row * 72 + col] = (_Float16)tanhf(acc[t][v]);
    }

  // ---------------- layer 1: h1 = tanh(h0 @ W1^T + b1) ----------------
#pragma unroll
  for (int t = 0; t < 4; t++) {
    float bv = b1[e * H_ + m + 16 * t];
#pragma unroll
    for (int v = 0; v < 8; v++) acc[t][v] = bv;
  }
#pragma unroll
  for (int c = 0; c < 2; c++) {
    v16h a = {};
#pragma unroll
    for (int h = 0; h < 16; h++) a[h] = myH0[m * 72 + c * 32 + a_klocal(h, hi)];
#pragma unroll
    for (int t = 0; t < 4; t++) {
      v16h bf = {};
#pragma unroll
      for (int j = 0; j < 16; j++) {
        int k = c * 32 + b_klocal(j, hi);
        bf[j] = (_Float16)sW1[(m + 16 * t) * 66 + k];     // W1[n][k] (B=W1^T)
      }
      acc[t] = __builtin_amdgcn_wmma_f32_16x16x32_f16(
          false, a, false, bf, (short)0, acc[t], false, false);
    }
  }
#pragma unroll
  for (int t = 0; t < 4; t++)
#pragma unroll
    for (int v = 0; v < 8; v++) {
      int row = v + 8 * hi, col = m + 16 * t;
      myG[row * 72 + col] = (_Float16)tanhf(acc[t][v]);
    }

  // ------- head: out = h1 . W2 + b2 ; g1 = W2 * (1 - h1^2) -------
  {
    int hb = hi * 32;
    float partial = 0.f;
#pragma unroll
    for (int h = 0; h < 32; h++)
      partial += (float)myG[m * 72 + hb + h] * sW2[hb + h];
    partial += __shfl_xor(partial, 16);
    float outv = partial + b2[e];
    if (hi == 0 && aRow < N_)
      atomicAdd(&energy[idx[e * N_ + aRow]], outv);
#pragma unroll
    for (int h = 0; h < 32; h++) {
      float h1v = (float)myG[m * 72 + hb + h];
      myG[m * 72 + hb + h] = (_Float16)(sW2[hb + h] * (1.f - h1v * h1v));
    }
  }

  // ------- backward: t0 = g1 @ W1 ; g0 = t0 * (1 - h0^2) -------
#pragma unroll
  for (int t = 0; t < 4; t++)
#pragma unroll
    for (int v = 0; v < 8; v++) acc[t][v] = 0.f;
#pragma unroll
  for (int c = 0; c < 2; c++) {
    v16h a = {};
#pragma unroll
    for (int h = 0; h < 16; h++) a[h] = myG[m * 72 + c * 32 + a_klocal(h, hi)];
#pragma unroll
    for (int t = 0; t < 4; t++) {
      v16h bf = {};
#pragma unroll
      for (int j = 0; j < 16; j++) {
        int k = c * 32 + b_klocal(j, hi);
        bf[j] = (_Float16)sW1[k * 66 + (m + 16 * t)];     // W1[k][n] (B=W1)
      }
      acc[t] = __builtin_amdgcn_wmma_f32_16x16x32_f16(
          false, a, false, bf, (short)0, acc[t], false, false);
    }
  }
#pragma unroll
  for (int t = 0; t < 4; t++)
#pragma unroll
    for (int v = 0; v < 8; v++) {
      int row = v + 8 * hi, col = m + 16 * t;
      float h0v = (float)myH0[row * 72 + col];
      myG[row * 72 + col] = (_Float16)(acc[t][v] * (1.f - h0v * h0v));
    }

  // ------- dE/dx = g0 @ W0  (N = P = 36 -> 3 n-tiles) -------
  v8f acc3[3];
#pragma unroll
  for (int t = 0; t < 3; t++)
#pragma unroll
    for (int v = 0; v < 8; v++) acc3[t][v] = 0.f;
#pragma unroll
  for (int c = 0; c < 2; c++) {
    v16h a = {};
#pragma unroll
    for (int h = 0; h < 16; h++) a[h] = myG[m * 72 + c * 32 + a_klocal(h, hi)];
#pragma unroll
    for (int t = 0; t < 3; t++) {
      v16h bf = {};
#pragma unroll
      for (int j = 0; j < 16; j++) {
        int k = c * 32 + b_klocal(j, hi);                 // h index
        int p = m + 16 * t;
        bf[j] = (_Float16)((p < P_) ? sW0[k * P_ + p] : 0.f);
      }
      acc3[t] = __builtin_amdgcn_wmma_f32_16x16x32_f16(
          false, a, false, bf, (short)0, acc3[t], false, false);
    }
  }
#pragma unroll
  for (int t = 0; t < 3; t++)
#pragma unroll
    for (int v = 0; v < 8; v++) {
      int row = v + 8 * hi, p = m + 16 * t;
      int a2 = tileAtom + row;
      if (p < P_ && a2 < N_) dE[(e * N_ + a2) * P_ + p] = acc3[t][v];
    }
}

// =====================================================================
// Stable counting sort of atoms by image index (build `order`).
// =====================================================================
__global__ void zero_kernel(float* out, int nout, int* hist, int nhist) {
  int i = blockIdx.x * blockDim.x + threadIdx.x;
  if (i < nout) out[i] = 0.f;
  if (i < nhist) hist[i] = 0;
}

__global__ void hist_kernel(const int* __restrict__ idx, int* __restrict__ hist) {
  int chunk = blockIdx.x;
  for (int t = threadIdx.x; t < CH_; t += 256) {
    int i = chunk * CH_ + t;
    if (i < Q_) atomicAdd(&hist[chunk * B_ + idx[i]], 1);
  }
}

// per-image exclusive scan over chunks; hist becomes chunk-exclusive offsets
__global__ void scan_chunks(int* __restrict__ hist, int* __restrict__ totals) {
  int b = blockIdx.x * blockDim.x + threadIdx.x;
  if (b >= B_) return;
  int run = 0;
  for (int c = 0; c < NCHUNK_; c++) {
    int t = hist[c * B_ + b];
    hist[c * B_ + b] = run;
    run += t;
  }
  totals[b] = run;
}

// exclusive scan of 2000 image totals -> image base offsets (single block)
__global__ void scan_images(const int* __restrict__ totals, int* __restrict__ imgBase) {
  __shared__ int segsum[256];
  int t = threadIdx.x, beg = t * 8, run = 0;
  int local[8];
#pragma unroll
  for (int i = 0; i < 8; i++) {
    int g = beg + i;
    int v = (g < B_) ? totals[g] : 0;
    local[i] = run;
    run += v;
  }
  segsum[t] = run;
  __syncthreads();
  if (t == 0) {
    int r = 0;
    for (int i = 0; i < 256; i++) { int v = segsum[i]; segsum[i] = r; r += v; }
  }
  __syncthreads();
  int basev = segsum[t];
#pragma unroll
  for (int i = 0; i < 8; i++) {
    int g = beg + i;
    if (g < B_) imgBase[g] = basev + local[i];
  }
}

__global__ void rank_kernel(const int* __restrict__ idx, const int* __restrict__ hist,
                            const int* __restrict__ imgBase, int* __restrict__ order) {
  __shared__ int sidx[CH_];
  int chunk = blockIdx.x, base = chunk * CH_;
  for (int t = threadIdx.x; t < CH_; t += 256) {
    int i = base + t;
    sidx[t] = (i < Q_) ? idx[i] : -1;
  }
  __syncthreads();
  for (int t = threadIdx.x; t < CH_; t += 256) {
    int i = base + t;
    if (i >= Q_) continue;
    int my = sidx[t], r = 0;
    for (int j = 0; j < t; j++) r += (sidx[j] == my);
    order[imgBase[my] + hist[chunk * B_ + my] + r] = i;
  }
}

// =====================================================================
// COO: force[col] -= val * dE[order[row/P] * P + row%P]
// =====================================================================
__global__ void sparse_kernel(const float* __restrict__ fp_vals,
                              const int* __restrict__ fp_rows,
                              const int* __restrict__ fp_cols,
                              const int* __restrict__ order,
                              const float* __restrict__ dE,
                              float* __restrict__ force) {
  int i = blockIdx.x * blockDim.x + threadIdx.x;
  if (i >= NNZ_) return;
  int r = fp_rows[i];
  int s = r / P_;
  int p = r - s * P_;
  int atom = order[s];
  float v = fp_vals[i] * dE[atom * P_ + p];
  atomicAdd(&force[fp_cols[i]], -v);
}

extern "C" void kernel_launch(void* const* d_in, const int* in_sizes, int n_in,
                              void* d_out, int out_size, void* d_ws, size_t ws_size,
                              hipStream_t stream) {
  (void)in_sizes; (void)n_in; (void)out_size; (void)ws_size;
  const float* x       = (const float*)d_in[0];
  const float* W0      = (const float*)d_in[1];
  const float* b0      = (const float*)d_in[2];
  const float* W1      = (const float*)d_in[3];
  const float* b1      = (const float*)d_in[4];
  const float* W2      = (const float*)d_in[5];
  const float* b2      = (const float*)d_in[6];
  const float* fp_vals = (const float*)d_in[7];
  const int*   idx     = (const int*)d_in[8];
  const int*   fp_rows = (const int*)d_in[9];
  const int*   fp_cols = (const int*)d_in[10];

  float* energy = (float*)d_out;        // [B_]
  float* force  = (float*)d_out + B_;   // [3*Q_]

  char* ws = (char*)d_ws;
  float* dE      = (float*)ws;                              // Q_*P_ floats
  int*   order   = (int*)(ws + (size_t)Q_ * P_ * sizeof(float));
  int*   hist    = order + Q_;                              // NCHUNK_*B_
  int*   totals  = hist + (size_t)NCHUNK_ * B_;             // B_
  int*   imgBase = totals + B_;                             // B_

  const int nout  = B_ + 3 * Q_;
  const int nhist = NCHUNK_ * B_;
  int zg = (nout > nhist ? nout : nhist);
  zero_kernel<<<(zg + 255) / 256, 256, 0, stream>>>(energy, nout, hist, nhist);

  mlp_fwd_bwd<<<dim3(TILES_PER_E_, E_), 256, 0, stream>>>(
      x, W0, b0, W1, b1, W2, b2, idx, energy, dE);

  hist_kernel<<<NCHUNK_, 256, 0, stream>>>(idx, hist);
  scan_chunks<<<(B_ + 255) / 256, 256, 0, stream>>>(hist, totals);
  scan_images<<<1, 256, 0, stream>>>(totals, imgBase);
  rank_kernel<<<NCHUNK_, 256, 0, stream>>>(idx, hist, imgBase, order);

  sparse_kernel<<<(NNZ_ + 255) / 256, 256, 0, stream>>>(
      fp_vals, fp_rows, fp_cols, order, dE, force);
}
